// eidosNeighborMixer_69741678952857
// MI455X (gfx1250) — matrix operand order repaired and checked by reading, not compiled, not run on hardware
//
#include <hip/hip_runtime.h>
#include <cmath>

#define D_MODEL 1024
#define KTOT    3072
#define SEQ     8192
#define MTOT    32768   // 4 * 8192 rows
#define NSTEP   96      // KTOT / 32
#define MT      64      // rows per block (four 16-row WMMA M-tiles)
#define AP      40      // padded A-tile pitch (bf16 elems), 32 + 8
#define ABUF    (MT * AP)
#define YP      1032    // padded y-tile pitch (f32 elems)
#define GROWS   8194    // guard-padded rows per batch: 1 + 8192 + 1
#define XGROWS  (4 * GROWS)
#define EPS_N   1e-6f

typedef __attribute__((ext_vector_type(16))) __bf16          v16bf;
typedef __attribute__((ext_vector_type(8)))  float           v8f;
typedef __attribute__((ext_vector_type(8)))  unsigned short  ush8;
typedef __attribute__((ext_vector_type(16))) unsigned short  ush16;
typedef __attribute__((ext_vector_type(4)))  unsigned int    v4u;
typedef __attribute__((ext_vector_type(4)))  int             v4i;
typedef __attribute__((ext_vector_type(8)))  int             v8i;

__device__ __forceinline__ unsigned short f2bf(float f) {
  unsigned u = __builtin_bit_cast(unsigned, f);
  u += 0x7FFFu + ((u >> 16) & 1u);   // round-to-nearest-even
  return (unsigned short)(u >> 16);
}

// ---- TDM: issue a 2D tensor_load_to_lds of a 64-row x 32-elem bf16 tile ----
// LDS pad: after every 16 DWORDs (one 32-elem row = 64B) insert 4 DWORDs (16B)
// -> LDS pitch = 80B = AP(40) bf16 elems, matching the A-tile layout.
__device__ __forceinline__ void tdm_load_tile(unsigned lds_off,
                                              const unsigned short* gptr) {
  unsigned long long ga = (unsigned long long)(uintptr_t)gptr;
  v4u g0;
  g0[0] = 1u;                                            // count=1, user mode
  g0[1] = lds_off;                                       // lds_addr (bytes)
  g0[2] = (unsigned)ga;                                  // global_addr[31:0]
  g0[3] = (unsigned)((ga >> 32) & 0x01FFFFFFu) | (2u << 30);  // addr[56:32] | type=2
  v8i g1;
  g1[0] = (int)((1u << 16)        // data_size = 2 bytes
              | (1u << 20)        // pad_enable
              | (3u << 22)        // pad_interval: 16 DWORDs
              | (3u << 25));      // pad_amount: 4 DWORDs
  g1[1] = 0;                      // abar addr / tensor_dim0[15:0] (dim0=1<<30)
  g1[2] = 0x4000;                 // tensor_dim0[31:16] = 1<<30 >> 16
  g1[3] = 0x4000 | (32 << 16);    // tensor_dim1 hi (1<<30) | tile_dim0 = 32
  g1[4] = 64;                     // tile_dim1 = 64 rows (tile_dim2 = 0)
  g1[5] = 1024;                   // tensor_dim0_stride = 1024 elems
  g1[6] = 0;
  g1[7] = 0;
  v4i gz = v4i{0, 0, 0, 0};
#if __has_include(<hip/amd_detail/amd_gfx1250_TDM.h>)
  v8i gz8 = v8i{0, 0, 0, 0, 0, 0, 0, 0};
  __builtin_amdgcn_tensor_load_to_lds(g0, g1, gz, gz, gz8, 0);
#else
  __builtin_amdgcn_tensor_load_to_lds(g0, g1, gz, gz, 0);
#endif
}

// Convert x [4][8192][1024] fp32 -> guard-padded bf16 xg [4][8194][1024]
// (zero row before and after each batch, so triplet shifts need no branches).
__global__ void conv_x_kernel(const float* __restrict__ x,
                              unsigned short* __restrict__ xg) {
  int g = blockIdx.x;            // padded row 0..XGROWS-1
  int c = threadIdx.x * 4;
  int batch = g / GROWS;
  int rr    = g - batch * GROWS;
  unsigned short* dst = xg + (size_t)g * D_MODEL + c;
  if (rr == 0 || rr == GROWS - 1) {
    dst[0] = 0; dst[1] = 0; dst[2] = 0; dst[3] = 0;
  } else {
    const float* src = x + (size_t)(batch * SEQ + rr - 1) * D_MODEL + c;
    float4 t = *(const float4*)src;
    dst[0] = f2bf(t.x); dst[1] = f2bf(t.y);
    dst[2] = f2bf(t.z); dst[3] = f2bf(t.w);
  }
}

// Pack W [K=3072, N=1024] fp32 into bf16 in WMMA-B per-lane order.
__global__ void pack_w_kernel(const float* __restrict__ W,
                              unsigned short* __restrict__ Wp) {
  int fid  = blockIdx.x * blockDim.x + threadIdx.x;   // < 3072*1024
  int j    = fid & 15;
  int lane = (fid >> 4) & 31;
  int nt   = (fid >> 9) & 63;
  int kb   = fid >> 15;
  int k = kb * 32 + ((j >> 3) << 4) + ((lane >> 4) << 3) + (j & 7);
  int n = nt * 16 + (lane & 15);
  Wp[fid] = f2bf(W[(size_t)k * D_MODEL + n]);
}

// Fold the two rotation planes: angle = theta[0][i] + theta[1][i]
__global__ void trig_kernel(const float* __restrict__ theta,
                            float* __restrict__ ct, float* __restrict__ st) {
  int i = threadIdx.x;   // 512 threads
  float a = theta[i] + theta[512 + i];
  ct[i] = cosf(a);
  st[i] = sinf(a);
}

__global__ void __launch_bounds__(256)
gemm_fused_kernel(const unsigned short* __restrict__ xg,
                  const unsigned short* __restrict__ Wp,
                  const float* __restrict__ bias,
                  const float* __restrict__ ct,
                  const float* __restrict__ st,
                  const float* __restrict__ gamma,
                  float* __restrict__ out) {
  __shared__ float          ytile[8 * YP];     // 33 KB, one 8-row slab at a time
  __shared__ unsigned short atile[3 * ABUF];   // triple-buffered A staging (bf16, TDM-filled)
  __shared__ float          gsum[64];          // 8 rows x 7 groups (padded to 8)

  const int tid     = threadIdx.x;
  const int lane    = tid & 31;
  const int wave    = tid >> 5;
  const int r16     = lane & 15;
  const int hi      = lane >> 4;
  const int rowBase = blockIdx.x * MT;
  const int batch   = rowBase >> 13;
  const int s0      = rowBase & (SEQ - 1);     // block never straddles a batch
  const unsigned ldsA = (unsigned)(uintptr_t)&atile[0];

  v8f acc0[8], acc1[8], acc2[8], acc3[8];
#pragma unroll
  for (int q = 0; q < 8; ++q) {
    acc0[q] = v8f{0.f, 0.f, 0.f, 0.f, 0.f, 0.f, 0.f, 0.f};
    acc1[q] = v8f{0.f, 0.f, 0.f, 0.f, 0.f, 0.f, 0.f, 0.f};
    acc2[q] = v8f{0.f, 0.f, 0.f, 0.f, 0.f, 0.f, 0.f, 0.f};
    acc3[q] = v8f{0.f, 0.f, 0.f, 0.f, 0.f, 0.f, 0.f, 0.f};
  }

  // Source pointer for k-step's A tile: 64 consecutive guard-padded rows.
  auto srcA = [&](int kstep) -> const unsigned short* {
    int seg = kstep >> 5;              // 0: left, 1: center, 2: right
    int d0  = (kstep & 31) << 5;
    return xg + ((size_t)(batch * GROWS + s0 + seg) * D_MODEL + d0);
  };

  // Two TDM loads in flight from the start (triple-buffered A tile).
  if (wave == 0) {
    tdm_load_tile(ldsA, srcA(0));
    tdm_load_tile(ldsA + (unsigned)(ABUF * 2), srcA(1));
  }

  for (int k = 0; k < NSTEP; ++k) {
    // Ensure tile k is resident (no-op for waves with TENSORcnt==0).
    if (k + 1 < NSTEP) __builtin_amdgcn_s_wait_tensorcnt(1);
    else               __builtin_amdgcn_s_wait_tensorcnt(0);
    __syncthreads();

    // Keep two DMA transfers in flight: issue tile k+2 now; its buffer's last
    // readers (compute of step k-1) all passed the barrier above.
    if (k + 2 < NSTEP && wave == 0)
      tdm_load_tile(ldsA + (unsigned)(((k + 2) % 3) * ABUF * 2), srcA(k + 2));

    const int cur = k % 3;

    // A fragments: lane holds row (lane&15) of each 16-row M-tile;
    // K chunks [hi*8..+8) and [16+hi*8..+8)
    const unsigned short* ap0 = atile + cur * ABUF + r16 * AP + hi * 8;
    v16bf af[4];
#pragma unroll
    for (int mt = 0; mt < 4; ++mt) {
      const unsigned short* ap = ap0 + mt * 16 * AP;
      ush8 alo = *(const ush8*)(ap);
      ush8 ahi = *(const ush8*)(ap + 16);
      af[mt] = __builtin_bit_cast(
          v16bf, __builtin_shufflevector(alo, ahi, 0, 1, 2, 3, 4, 5, 6, 7,
                                         8, 9, 10, 11, 12, 13, 14, 15));
    }

    const ush16* wp = (const ush16*)Wp + ((size_t)k * 64 + wave * 8) * 32 + lane;

    // Prefetch next k-step's 8 KB B slice toward the WGP-level cache.
    if (k + 1 < NSTEP) {
      const char* pf = (const char*)((const ush16*)Wp +
                                     ((size_t)(k + 1) * 64 + wave * 8) * 32) +
                       lane * 256;
      __builtin_prefetch(pf, 0, 3);
    }

    // Software-pipeline the B fragment by one n-tile so each load's wait sits
    // behind four WMMAs instead of in front of them.
    v16bf bcur = __builtin_bit_cast(v16bf, wp[0]);
#pragma unroll
    for (int q = 0; q < 8; ++q) {
      v16bf bnxt;
      if (q < 7) bnxt = __builtin_bit_cast(v16bf, wp[(q + 1) * 32]);
      acc0[q] = __builtin_amdgcn_wmma_f32_16x16x32_bf16(
          false, af[0], false, bcur, (short)0, acc0[q], false, false);
      acc1[q] = __builtin_amdgcn_wmma_f32_16x16x32_bf16(
          false, af[1], false, bcur, (short)0, acc1[q], false, false);
      acc2[q] = __builtin_amdgcn_wmma_f32_16x16x32_bf16(
          false, af[2], false, bcur, (short)0, acc2[q], false, false);
      acc3[q] = __builtin_amdgcn_wmma_f32_16x16x32_bf16(
          false, af[3], false, bcur, (short)0, acc3[q], false, false);
      if (q < 7) bcur = bnxt;
    }
  }
  __syncthreads();

  // ---- Fused epilogue: bias + rotation + mod-7 RMS norm + gamma ----
  // Eight 8-row slabs (unrolled so accumulator indexing stays static).
#pragma unroll
  for (int half = 0; half < 8; ++half) {
    const int mt   = half >> 1;        // which 16-row M-tile
    const int hsel = half & 1;         // which lane half holds these rows
    if (hi == hsel) {
#pragma unroll
      for (int q = 0; q < 8; ++q) {
        int col  = (wave * 8 + q) * 16 + r16;
        float bb = bias[col];
#pragma unroll
        for (int v = 0; v < 8; ++v) {
          float a = (mt == 0) ? acc0[q][v]
                  : (mt == 1) ? acc1[q][v]
                  : (mt == 2) ? acc2[q][v]
                              : acc3[q][v];
          ytile[v * YP + col] = a + bb;
        }
      }
    }
    if (tid < 64) gsum[tid] = 0.f;
    __syncthreads();

    // Rotate pairs and accumulate per-(row, d%7) square sums.
    {
      int r = tid >> 5;   // 0..7 (one wave per row)
      float ls[7];
#pragma unroll
      for (int g = 0; g < 7; ++g) ls[g] = 0.f;
      for (int j = 0; j < 16; ++j) {
        int i  = (tid & 31) + (j << 5);   // pair index 0..511
        int c0 = i << 1;
        float a  = ytile[r * YP + c0];
        float cc = ytile[r * YP + c0 + 1];
        float cv = ct[i], sv = st[i];
        float ra = a * cv - cc * sv;
        float rb = a * sv + cc * cv;
        ytile[r * YP + c0]     = ra;
        ytile[r * YP + c0 + 1] = rb;
        int g0 = c0 % 7;
        int g1 = (c0 + 1) % 7;
        float sq0 = ra * ra, sq1 = rb * rb;
#pragma unroll
        for (int g = 0; g < 7; ++g)
          ls[g] += (g == g0 ? sq0 : 0.f) + (g == g1 ? sq1 : 0.f);
      }
#pragma unroll
      for (int g = 0; g < 7; ++g) atomicAdd(&gsum[r * 8 + g], ls[g]);
    }
    __syncthreads();

    if (tid < 56) {
      int r = tid / 7, g = tid % 7;
      float cnt = (g < 2) ? 147.0f : 146.0f;   // 1024 = 146*7 + 2
      float ss  = gsum[r * 8 + g];
      gsum[r * 8 + g] = 1.0f / sqrtf(ss / cnt + EPS_N);
    }
    __syncthreads();

    // Normalize, apply gamma, store fp32.
    for (int j = 0; j < 32; ++j) {
      int idx = tid + (j << 8);
      int r   = idx >> 10;
      int c   = idx & 1023;
      out[(size_t)(rowBase + half * 8 + r) * D_MODEL + c] =
          ytile[r * YP + c] * gsum[r * 8 + c % 7] * gamma[c];
    }
    __syncthreads();
  }
}

extern "C" void kernel_launch(void* const* d_in, const int* in_sizes, int n_in,
                              void* d_out, int out_size, void* d_ws, size_t ws_size,
                              hipStream_t stream) {
  (void)in_sizes; (void)n_in; (void)out_size; (void)ws_size;
  const float* x     = (const float*)d_in[0];
  const float* W     = (const float*)d_in[1];
  const float* b     = (const float*)d_in[2];
  const float* theta = (const float*)d_in[3];
  const float* gamma = (const float*)d_in[4];
  float* out = (float*)d_out;

  unsigned short* xg = (unsigned short*)d_ws;                    // 67 MB guard-padded bf16 x
  size_t xg_bytes = (size_t)XGROWS * D_MODEL * 2;
  unsigned short* Wp = (unsigned short*)((char*)d_ws + xg_bytes);// 6 MB bf16 packed W
  float* ct = (float*)((char*)Wp + (size_t)KTOT * D_MODEL * 2);  // 512 f32
  float* st = ct + 512;                                          // 512 f32

  conv_x_kernel<<<XGROWS, 256, 0, stream>>>(x, xg);
  pack_w_kernel<<<(KTOT * D_MODEL) / 256, 256, 0, stream>>>(W, Wp);
  trig_kernel<<<1, 512, 0, stream>>>(theta, ct, st);
  gemm_fused_kernel<<<MTOT / MT, 256, 0, stream>>>(xg, Wp, b, ct, st, gamma, out);
}
